// StrideConv_53034256171738
// MI455X (gfx1250) — matrix operand order repaired and checked by reading, not compiled, not run on hardware
//
#include <hip/hip_runtime.h>

// ---------------- problem constants ----------------
#define CIN   64
#define COUT  64
#define RIN   6
#define ROUT  6
#define CHARTS 5
#define HH    64
#define WW    128
#define BATCH 4

#define MTOT  (COUT*ROUT)          // 384
#define KC    (CIN*RIN)            // 384 channels per tap
#define NROWS (BATCH*CHARTS*HH)    // 1280 row-blocks (each = 128 cols)
#define SLAB  (RIN*CHARTS*HH*WW)   // 245760 elems per (b,ci)
#define KCHUNKS 12                 // 384/32
#define MTILES  24                 // 384/16
// A fragment image: [tap 9][kchunk 12][mtile 24][lane 32][16 halves]
#define AFRAG_HALVES (9*KCHUNKS*MTILES*32*16)   // 1,327,104

typedef __bf16 v16bf __attribute__((ext_vector_type(16)));
typedef float  v8f   __attribute__((ext_vector_type(8)));
typedef unsigned int uint4v __attribute__((ext_vector_type(4)));

// _IDX_K % 7 (rows = ro, cols = tap kh*3+kw)
__constant__ int c_idxk[54] = {
  5,4,6,6,0,3,6,1,2,
  4,3,6,5,0,2,6,6,1,
  3,2,6,4,0,1,6,5,6,
  2,1,6,3,0,6,6,4,5,
  1,6,6,2,0,5,6,3,4,
  6,5,6,1,0,4,6,2,3 };

// ------------------------------------------------------------------
// Halo-gather source mapping: replicates _build_reorder_idx for r=6.
// Returns element offset into the per-(b,ci) slab (RIN,5,64,128),
// or -1 if the source is VMASK-killed. Unfilled positions gather idx 0.
// ------------------------------------------------------------------
__device__ __forceinline__ int src_off(int r, int c, int yy, int xx) {
  int rr, cc, y, x;
  if (yy >= 1 && yy <= 64 && xx >= 1 && xx <= 128) {           // interior
    rr = r; cc = c; y = yy - 1; x = xx - 1;
  } else if (yy == 0) {
    if (xx >= 1 && xx <= 64)       { rr = r;         cc = (c+4)%5; y = 63;          x = 64 + (xx-1); }
    else if (xx >= 65 && xx <= 128){ int k = xx-65;  rr = (r+1)%6; cc = (c+4)%5; y = 63-k; x = 127; }
    else return 0;
  } else if (yy == 65) {
    if (xx >= 65 && xx <= 128)     { int k = xx-65;  rr = r;       cc = (c+1)%5; y = 0;    x = k;  }
    else if (xx >= 2 && xx <= 64)  { int k = xx-2;   rr = (r+1)%6; cc = (c+1)%5; y = 63-k; x = 0;  }
    else return 0;
  } else if (xx == 0) {             // yy in [1,64]
    int k = yy-1; rr = (r+5)%6; cc = (c+4)%5; y = 63; x = 63-k;
  } else if (xx == 129) {
    if (yy >= 2) { int k = yy-2; rr = (r+5)%6; cc = (c+1)%5; y = 0; x = 127-k; }
    else return 0;
  } else return 0;
  if ((y == 0 && x == 127) || (y == 63 && x == 0)) return -1;  // VMASK
  return ((rr*5 + cc)*64 + y)*128 + x;
}

// ------------------------------------------------------------------
// Kernel 1: expand hex weights into bf16 hi/lo A-fragment images.
// One thread per dword (pair of halves).
// ------------------------------------------------------------------
__global__ __launch_bounds__(256)
void expand_weights(const float* __restrict__ weight,
                    __bf16* __restrict__ Ahi, __bf16* __restrict__ Alo) {
  int idx = blockIdx.x * blockDim.x + threadIdx.x;
  if (idx >= 9*KCHUNKS*MTILES*32*8) return;
  int d    = idx & 7;
  int lane = (idx >> 3) & 31;
  int rem  = idx >> 8;           // (t*12+kk)*24 + mt
  int mt   = rem % MTILES;
  int tkk  = rem / MTILES;
  int kk   = tkk % KCHUNKS;
  int t    = tkk / KCHUNKS;
  int kh = t / 3, kw = t % 3;
  bool masked = (kh == 0 && kw == 2) || (kh == 2 && kw == 0);
  int m  = mt*16 + (lane & 15);
  int co = m / 6, ro = m - co*6;
  // k order used here (same as B): lanes<16: K 0..15; lanes>=16: K 16..31
  int kbase = ((lane >> 4) ? 16 : 0) + 2*d;
  __bf16 h2[2], l2[2];
  #pragma unroll
  for (int hs = 0; hs < 2; ++hs) {
    int kc = kk*32 + kbase + hs;
    int ci = kc / 6, ri = kc - ci*6;
    float w = 0.f;
    if (!masked) {
      int rmap = ri - ro; if (rmap < 0) rmap += 6;
      w = weight[((co*CIN + ci)*RIN + rmap)*7 + c_idxk[ro*9 + t]];
    }
    __bf16 h = (__bf16)w;
    h2[hs] = h;
    l2[hs] = (__bf16)(w - (float)h);
  }
  Ahi[2*idx]   = h2[0]; Ahi[2*idx+1] = h2[1];
  Alo[2*idx]   = l2[0]; Alo[2*idx+1] = l2[1];
}

// ------------------------------------------------------------------
// Kernel 2: fused halo-gather + implicit GEMM, bf16x3 WMMA.
// grid.x = 1280 (b,chart,i) row blocks; grid.y = 6 (M blocks of 64).
// 256 threads = 8 waves; wave grid 2(M) x 4(N); 32x32 per wave.
// ------------------------------------------------------------------
__global__ __launch_bounds__(256)
void ico_conv_wmma(const float* __restrict__ x,
                   const __bf16* __restrict__ Ahi,
                   const __bf16* __restrict__ Alo,
                   const float* __restrict__ bias,
                   float* __restrict__ out) {
  __shared__ int     offs[6][128];
  __shared__ __bf16  Bhi[8][32][16];
  __shared__ __bf16  Blo[8][32][16];

  const int tid   = threadIdx.x;
  const int lane  = tid & 31;
  const int wave  = tid >> 5;
  const int waveM = wave >> 2;         // 0..1
  const int waveN = wave & 3;          // 0..3
  const int rid   = blockIdx.x;        // (b,chart,i)
  const int mb    = blockIdx.y;        // M block of 64
  const int b     = rid / (CHARTS*HH);
  const int rem   = rid - b*(CHARTS*HH);
  const int chart = rem >> 6;
  const int i     = rem & 63;

  const float* xb = x + (size_t)b * CIN * SLAB;

  // B-loader thread mapping: tid <-> (ntile, lane-image slot)
  const int nt8  = tid >> 5;           // 0..7: which 16-col N tile
  const int nloc = lane & 15;
  const int n    = nt8*16 + nloc;      // j column this thread gathers
  const int koff = (lane < 16) ? 0 : 16;

  v8f acc[2][2];
  v8f z = {};
  acc[0][0] = z; acc[0][1] = z; acc[1][0] = z; acc[1][1] = z;

  for (int t = 0; t < 9; ++t) {
    if (t == 2 || t == 6) continue;    // MASK3 zeros these taps
    const int kh = t / 3, kw = t % 3;

    // per-tap halo offset table (768 entries)
    for (int q = tid; q < 6*128; q += 256) {
      int ri = q >> 7, jj = q & 127;
      offs[ri][jj] = src_off(ri, chart, i + kh, jj + kw);
    }
    __syncthreads();

    for (int kk = 0; kk < KCHUNKS; ++kk) {
      // ---- gather + split to bf16 hi/lo, write lane-image to LDS ----
      union { __bf16 h[16]; uint4v q[2]; } uh, ul;
      #pragma unroll
      for (int u = 0; u < 16; ++u) {
        int kc = kk*32 + koff + u;     // k order used here (same as A): lanes<16 K 0..15, else 16..31
        int ci = kc / 6;
        int ri = kc - ci*6;
        int off = offs[ri][n];
        float v = (off < 0) ? 0.f : xb[ci*SLAB + off];
        __bf16 h = (__bf16)v;
        uh.h[u] = h;
        ul.h[u] = (__bf16)(v - (float)h);
      }
      *reinterpret_cast<uint4v*>(&Bhi[nt8][lane][0]) = uh.q[0];
      *reinterpret_cast<uint4v*>(&Bhi[nt8][lane][8]) = uh.q[1];
      *reinterpret_cast<uint4v*>(&Blo[nt8][lane][0]) = ul.q[0];
      *reinterpret_cast<uint4v*>(&Blo[nt8][lane][8]) = ul.q[1];
      __syncthreads();

      // ---- load fragments ----
      v16bf ahi[2], alo[2], bhiF[2], bloF[2];
      #pragma unroll
      for (int ms = 0; ms < 2; ++ms) {
        int mt = mb*4 + waveM*2 + ms;
        size_t aoff = ((((size_t)t*KCHUNKS + kk)*MTILES + mt)*32 + lane) * 16;
        ahi[ms] = *reinterpret_cast<const v16bf*>(Ahi + aoff);
        alo[ms] = *reinterpret_cast<const v16bf*>(Alo + aoff);
      }
      #pragma unroll
      for (int ns = 0; ns < 2; ++ns) {
        int nt = waveN*2 + ns;
        bhiF[ns] = *reinterpret_cast<const v16bf*>(&Bhi[nt][lane][0]);
        bloF[ns] = *reinterpret_cast<const v16bf*>(&Blo[nt][lane][0]);
      }

      // ---- bf16x3 WMMA: hi*hi + hi*lo + lo*hi ----
      #pragma unroll
      for (int ms = 0; ms < 2; ++ms) {
        #pragma unroll
        for (int ns = 0; ns < 2; ++ns) {
          v8f a = acc[ms][ns];
          a = __builtin_amdgcn_wmma_f32_16x16x32_bf16(false, ahi[ms], false, bhiF[ns], (short)0, a, false, false);
          a = __builtin_amdgcn_wmma_f32_16x16x32_bf16(false, ahi[ms], false, bloF[ns], (short)0, a, false, false);
          a = __builtin_amdgcn_wmma_f32_16x16x32_bf16(false, alo[ms], false, bhiF[ns], (short)0, a, false, false);
          acc[ms][ns] = a;
        }
      }
      __syncthreads();
    }
  }

  // ---- epilogue: bias, VMASK, store (D row index == co*6+ro == m) ----
  const size_t outbase = (size_t)b * (MTOT*CHARTS*HH*WW/ROUT*ROUT/COUT*COUT) // b*15728640
                         ;
  const size_t ob = (size_t)b * ((size_t)MTOT*CHARTS*HH*WW) / 1 // placate
                    ;
  (void)outbase; (void)ob;
  const size_t base = (size_t)b * ((size_t)MTOT * CHARTS * HH * WW) / 1;
  const size_t rowoff = (size_t)chart * (HH*WW) + (size_t)i * WW;
  #pragma unroll
  for (int ms = 0; ms < 2; ++ms) {
    int mrow0 = mb*64 + (waveM*2 + ms)*16 + ((lane >> 4) << 3);
    #pragma unroll
    for (int ns = 0; ns < 2; ++ns) {
      int j = (waveN*2 + ns)*16 + (lane & 15);
      bool kill = (i == 0 && j == 127) || (i == 63 && j == 0);
      #pragma unroll
      for (int v = 0; v < 8; ++v) {
        int m  = mrow0 + v;
        int co = m / 6;
        float val = acc[ms][ns][v] + bias[co];
        if (kill) val = 0.f;
        out[base + (size_t)m * (CHARTS*HH*WW) + rowoff + j] = val;
      }
    }
  }
}

// ------------------------------------------------------------------
extern "C" void kernel_launch(void* const* d_in, const int* in_sizes, int n_in,
                              void* d_out, int out_size, void* d_ws, size_t ws_size,
                              hipStream_t stream) {
  (void)in_sizes; (void)n_in; (void)out_size; (void)ws_size;
  const float* x      = (const float*)d_in[0];
  const float* weight = (const float*)d_in[1];
  const float* bias   = (const float*)d_in[2];
  float*       out    = (float*)d_out;

  __bf16* Ahi = (__bf16*)d_ws;
  __bf16* Alo = Ahi + AFRAG_HALVES;

  const int total = 9*KCHUNKS*MTILES*32*8;      // 663,552 threads
  expand_weights<<<(total + 255)/256, 256, 0, stream>>>(weight, Ahi, Alo);

  dim3 grid(NROWS, MTOT/64, 1);                 // 1280 x 6
  ico_conv_wmma<<<grid, 256, 0, stream>>>(x, Ahi, Alo, bias, out);
}
